// LearnedOuterProd_26293789786402
// MI455X (gfx1250) — matrix operand order repaired and checked by reading, not compiled, not run on hardware
//
#include <hip/hip_runtime.h>

typedef __attribute__((ext_vector_type(16))) _Float16 v16h;
typedef __attribute__((ext_vector_type(8)))  _Float16 v8h;
typedef __attribute__((ext_vector_type(8)))  float    v8f;

#define BB_   2
#define NN_   512
#define DIN_  256
#define HH_   16
#define DOUT_ 128

// ---------------------------------------------------------------------------
// Kernel 1: LayerNorm + input projection.
// One block per row (b*512+n), 256 threads. Produces proj (f32, 32/row) and
// the 'a' half pre-converted to f16 for WMMA.
// ---------------------------------------------------------------------------
__global__ __launch_bounds__(256) void ln_proj_kernel(
    const float* __restrict__ feats, const float* __restrict__ gamma,
    const float* __restrict__ beta,  const float* __restrict__ W_in,
    const float* __restrict__ b_in,  float* __restrict__ proj_ws,
    _Float16* __restrict__ a_h)
{
    __shared__ float xs[DIN_];
    __shared__ float red[256];
    const int row = blockIdx.x;
    const int t   = threadIdx.x;

    float v = feats[row * DIN_ + t];

    red[t] = v;
    __syncthreads();
    for (int s = 128; s > 0; s >>= 1) { if (t < s) red[t] += red[t + s]; __syncthreads(); }
    const float mu = red[0] * (1.0f / DIN_);
    __syncthreads();

    const float d = v - mu;
    red[t] = d * d;
    __syncthreads();
    for (int s = 128; s > 0; s >>= 1) { if (t < s) red[t] += red[t + s]; __syncthreads(); }
    const float var  = red[0] * (1.0f / DIN_);
    const float rstd = rsqrtf(var + 1e-5f);

    xs[t] = d * rstd * gamma[t] + beta[t];
    __syncthreads();

    if (t < 32) {
        float acc = b_in[t];
        #pragma unroll 8
        for (int i = 0; i < DIN_; ++i) acc += xs[i] * W_in[i * 32 + t];
        proj_ws[row * 32 + t] = acc;
        if (t < HH_) a_h[row * HH_ + t] = (_Float16)acc;   // 'a' half
    }
}

// ---------------------------------------------------------------------------
// Kernel 2: K=32-padded T tensor:
//   Tp[b,m,d,k] : k<16  -> sum_q Wd[d,k,q] * bq[b,m,q]   (the real T)
//                 k==16 -> b_out[d]                      (bias row; A[.,16]=1)
//                 k>16  -> 0                             (zero padding)
// One block per (b*512+m), 256 threads, 16 slots each (128*32 = 4096/row).
// ---------------------------------------------------------------------------
__global__ __launch_bounds__(256) void make_T_kernel(
    const float* __restrict__ proj_ws, const float* __restrict__ W_out,
    const float* __restrict__ b_out,   _Float16* __restrict__ Tp)
{
    __shared__ float bqs[HH_];
    const int row = blockIdx.x;          // b*512 + m
    const int t   = threadIdx.x;
    if (t < HH_) bqs[t] = proj_ws[row * 32 + HH_ + t];   // 'bq' half
    __syncthreads();

    #pragma unroll
    for (int j = 0; j < 16; ++j) {
        const int idx = j * 256 + t;     // 0..4095 -> (d, k)
        const int dd  = idx >> 5;
        const int k   = idx & 31;
        float acc = 0.f;
        if (k < HH_) {
            const float* w = W_out + dd * 256 + k * 16;
            #pragma unroll
            for (int q = 0; q < 16; ++q) acc += w[q] * bqs[q];
        } else if (k == HH_) {
            acc = b_out[dd];
        }
        Tp[(size_t)row * (DOUT_ * 32) + idx] = (_Float16)acc;
    }
}

// ---------------------------------------------------------------------------
// Kernel 3: out[b,n,m,d] = sum_p a[b,n,p] * T[b,m,d,p] + b_out[d]
// One wave per (b, n-tile-of-16, m): 8 x v_wmma_f32_16x16x32_f16.
// Bias comes through the K=16 slot (A=1, B=b_out), so C starts at inline 0.
// All 32 lanes load B uniformly from the padded Tp -> no exec divergence,
// no per-iteration zeroing. NT dword stores (output 256MB > 192MB L2).
// ---------------------------------------------------------------------------
__global__ __launch_bounds__(256) void outer_wmma_kernel(
    const _Float16* __restrict__ a_h, const _Float16* __restrict__ Tp,
    float* __restrict__ out)
{
    const int lane = threadIdx.x & 31;
    const int wid  = blockIdx.x * 8 + (threadIdx.x >> 5);
    const int m    = wid % NN_;
    const int n16  = (wid / NN_) % (NN_ / 16);
    const int bb   = wid / (NN_ * (NN_ / 16));
    const int n0   = n16 * 16;

    // ---- A tile (16-bit A 16x32 layout):
    // lanes 0-15 : row M=lane,    elems 0..7 = K 0..7,  elems 8..15 = K 16..23
    // lanes 16-31: row M=lane-16, elems 0..7 = K 8..15, elems 8..15 = K 24..31
    // K=0..15 <- a[n,p]; K=16 <- 1.0 (bias row, lives in elem 8 of lanes 0-15).
    v16h A;
    {
        const int r = bb * NN_ + n0 + (lane & 15);
        const v8h av = *(const v8h*)(a_h + (size_t)r * HH_ + (lane >> 4) * 8);
        #pragma unroll
        for (int i = 0; i < 8; ++i) { A[i] = av[i]; A[8 + i] = (_Float16)0.f; }
        if (lane < 16) A[8] = (_Float16)1.0f;
    }

    // ---- B addressing: col N = lane&15 (-> d = d0 + N); K half = lane>>4.
    const _Float16* Tbase = Tp + (size_t)(bb * NN_ + m) * (DOUT_ * 32)
                               + (size_t)(lane & 15) * 32 + (size_t)(lane >> 4) * 8;
    float* obase = out + (((size_t)(bb * NN_ + n0)) * NN_ + m) * DOUT_ + (lane & 15);

    #pragma unroll 2
    for (int db = 0; db < 8; ++db) {
        const _Float16* tp = Tbase + db * 16 * 32;
        const v8h b0 = *(const v8h*)tp;
        const v8h b1 = *(const v8h*)(tp + 16);
        v16h Bv;
        #pragma unroll
        for (int i = 0; i < 8; ++i) { Bv[i] = b0[i]; Bv[8 + i] = b1[i]; }

        v8f c = {};   // inline-0 accumulator; bias arrives via K=16 slot
        c = __builtin_amdgcn_wmma_f32_16x16x32_f16(
                /*neg_a=*/false, A, /*neg_b=*/false, Bv,
                /*c_mod=*/(short)0, c, /*reuse_a=*/false, /*reuse_b=*/false);

        // VGPR r, lanes 0-15 -> row M=r; lanes 16-31 -> row M=r+8.
        #pragma unroll
        for (int r = 0; r < 8; ++r) {
            float* op = obase + (size_t)(r + ((lane >> 4) << 3)) * (NN_ * DOUT_) + db * 16;
            __builtin_nontemporal_store(c[r], op);
        }
    }
}

// ---------------------------------------------------------------------------
extern "C" void kernel_launch(void* const* d_in, const int* in_sizes, int n_in,
                              void* d_out, int out_size, void* d_ws, size_t ws_size,
                              hipStream_t stream) {
    const float* feats  = (const float*)d_in[0];
    const float* gamma  = (const float*)d_in[1];
    const float* beta   = (const float*)d_in[2];
    const float* W_in   = (const float*)d_in[3];
    const float* b_in   = (const float*)d_in[4];
    const float* W_out  = (const float*)d_in[5];
    const float* b_out  = (const float*)d_in[6];
    float* out = (float*)d_out;

    // Workspace layout (~8.4 MB):
    //   [0, 128KB)          proj f32  (1024 rows x 32)
    //   [128KB, 160KB)      a    f16  (1024 rows x 16)
    //   [160KB, 160KB+8MB)  Tp   f16  (1024 rows x 128 x 32, K-padded + bias row)
    char* ws = (char*)d_ws;
    float*    proj_ws = (float*)ws;
    _Float16* a_h     = (_Float16*)(ws + (size_t)1024 * 32 * 4);
    _Float16* Tp      = (_Float16*)(ws + (size_t)1024 * 32 * 4 + (size_t)1024 * 16 * 2);

    ln_proj_kernel   <<<1024, 256, 0, stream>>>(feats, gamma, beta, W_in, b_in, proj_ws, a_h);
    make_T_kernel    <<<1024, 256, 0, stream>>>(proj_ws, W_out, b_out, Tp);
    // waves = 2 * 32 * 512 = 32768 ; 8 waves / block -> 4096 blocks
    outer_wmma_kernel<<<4096, 256, 0, stream>>>(a_h, Tp, out);
}
